// SelfAttention_35459249996685
// MI455X (gfx1250) — matrix operand order repaired and checked by reading, not compiled, not run on hardware
//
#include <hip/hip_runtime.h>
#include <hip/hip_bf16.h>
#include <math.h>

// ---------------------------------------------------------------------------
// Self-attention (H=8, D=1024, L=2048) for MI455X (gfx1250, wave32, WMMA).
// All GEMM stages run on v_wmma_f32_16x16x32_bf16 (bf16 in, f32 accum).
// Tile staging uses the CDNA5 Tensor Data Mover (tensor_load_to_lds, one
// 8KB 128x32 tile per descriptor, TENSORcnt) for k-contiguous tiles, plus
// async global->LDS (ASYNCcnt) for the NT B-tile. Completion enforced with
// s_wait_tensorcnt / s_wait_asynccnt before the workgroup barrier.
// ---------------------------------------------------------------------------

#define H_  8
#define D_  1024
#define L_  2048
#define HD_ 8192

typedef __bf16 bf16;
typedef __attribute__((ext_vector_type(16))) bf16  v16bf;
typedef __attribute__((ext_vector_type(8)))  bf16  v8bf;
typedef __attribute__((ext_vector_type(4)))  bf16  v4bf;
typedef __attribute__((ext_vector_type(8)))  float v8f;
typedef __attribute__((ext_vector_type(4)))  unsigned v4u;
typedef __attribute__((ext_vector_type(8)))  unsigned v8u;

#define BM 128
#define BN 128
#define BK 32

static __device__ __forceinline__ bf16 f2bf(float f) {
  unsigned u = __builtin_bit_cast(unsigned, f);
  u += 0x7FFFu + ((u >> 16) & 1u);           // round-to-nearest-even
  unsigned short s = (unsigned short)(u >> 16);
  return __builtin_bit_cast(bf16, s);
}

// Generic LDS pointer -> 32-bit LDS byte address (flat aperture: addr[31:0]).
static __device__ __forceinline__ unsigned lds_addr(const void* p) {
  return (unsigned)(uintptr_t)p;
}

// TDM: DMA one [rows x 32] bf16 tile (row stride = lda elements) from global
// memory into LDS at ldsoff, packed [rows][32] (TDM walks X then Y, exactly
// our LDS tile layout). 2D descriptor: groups 2/3 omitted (NULL).
// D# per CDNA5 ISA.
static __device__ __forceinline__ void tdm_load_tile(unsigned ldsoff,
                                                     const bf16* g, long lda,
                                                     unsigned rows) {
  const unsigned long long ga = (unsigned long long)(uintptr_t)g;
  v4u g0;
  g0[0] = 1u;                                   // count=1 (valid user D#)
  g0[1] = ldsoff;                               // lds_addr (bytes)
  g0[2] = (unsigned)ga;                         // global_addr[31:0]
  g0[3] = (unsigned)(ga >> 32) | 0x80000000u;   // global_addr[56:32]|type=2
  v8u g1;
  g1[0] = 1u << 16;                             // data_size=1 (2 bytes)
  g1[1] = 0u;                                   // tensor_dim0[15:0] (=1<<20)
  g1[2] = 16u;                                  // td0[31:16]=16, td1[15:0]=0
  g1[3] = 16u | (32u << 16);                    // td1[31:16]=16, tile_dim0=32
  g1[4] = rows;                                 // tile_dim1=rows, tile_dim2=0
  g1[5] = (unsigned)lda;                        // tensor_dim0_stride[31:0]
  g1[6] = 0u;                                   // stride0[47:32], stride1 lo
  g1[7] = 0u;                                   // stride1 hi
  asm volatile("tensor_load_to_lds %0, %1" :: "s"(g0), "s"(g1) : "memory");
}

// 32 bytes per lane: cache/memory -> LDS without touching VGPRs (ASYNCcnt).
// INST_OFFSET is added to both the LDS and global addresses (ISA 15.18.3).
static __device__ __forceinline__ void async_copy_32B(unsigned lds, const void* g) {
  asm volatile("global_load_async_to_lds_b128 %0, %1, off\n\t"
               "global_load_async_to_lds_b128 %0, %1, off offset:16"
               :: "v"(lds), "v"(g) : "memory");
}

static __device__ __forceinline__ void wait_async() {
  asm volatile("s_wait_asynccnt 0" ::: "memory");
}

// A fragment (16x32 bf16, M x K): lane r=lane&15 holds row M=r; half=lane>>4
// selects K ranges {0..7,16..23} (half=0) or {8..15,24..31} (half=1).
static __device__ __forceinline__ v16bf frag_a(const bf16* Ts, int base, int lane) {
  const int half = lane >> 4, r = lane & 15;
  const bf16* p = Ts + (base + r) * BK + 8 * half;
  v8bf lo = *(const v8bf*)p;
  v8bf hi = *(const v8bf*)(p + 16);
  return __builtin_shufflevector(lo, hi, 0,1,2,3,4,5,6,7,8,9,10,11,12,13,14,15);
}

// B fragment (32x16 bf16, K x N): lane c=lane&15 holds col N=c; half selects
// K ranges {0..7,16..23} (half=0) or {8..15,24..31} (half=1). Bs stored [n][k].
static __device__ __forceinline__ v16bf frag_b(const bf16* Ts, int base, int lane) {
  const int half = lane >> 4, c = lane & 15;
  const bf16* p = Ts + (base + c) * BK + 8 * half;
  v8bf lo = *(const v8bf*)p;
  v8bf hi = *(const v8bf*)(p + 16);
  return __builtin_shufflevector(lo, hi, 0,1,2,3,4,5,6,7,8,9,10,11,12,13,14,15);
}

// ---------------- C = A(MxK, k-contig) @ B(KxN, n-contig) [+bias_m] -> bf16
__global__ __launch_bounds__(256)
void gemm_nn_bf16(const bf16* __restrict__ A, long lda,
                  const bf16* __restrict__ B, long ldb,
                  const float* __restrict__ bias,
                  bf16* __restrict__ C, long ldc, int K)
{
  __shared__ bf16 As[BM * BK];
  __shared__ bf16 Bs[BN * BK];
  const int t = threadIdx.x, lane = t & 31, wave = t >> 5;
  const int m0 = blockIdx.y * BM, n0 = blockIdx.x * BN;
  const int wm = (wave & 1) * 64, wn = (wave >> 1) * 32;
  v8f acc[4][2] = {};

  const int brow = t >> 3, bnp = (t & 7) * 16;     // B: 32 k-rows x 8 n-chunks
  const unsigned ldsAs = lds_addr(As);

  for (int k0 = 0; k0 < K; k0 += BK) {
    // A tile: one TDM descriptor DMAs the whole 128x32 tile into LDS
    if (wave == 0)
      tdm_load_tile(ldsAs, A + (size_t)m0 * lda + k0, lda, BM);
    {
      // B tile needs an n->k transpose: stage through VGPRs
      const bf16* src = B + (size_t)(k0 + brow) * ldb + n0 + bnp;
      v8bf b0 = ((const v8bf*)src)[0];
      v8bf b1 = ((const v8bf*)src)[1];
      if (k0 + BK < K) __builtin_prefetch(src + (size_t)BK * ldb, 0, 1);
#pragma unroll
      for (int j = 0; j < 8; ++j) {                // transpose into Bs[n][k]
        Bs[(bnp + j)     * BK + brow] = b0[j];
        Bs[(bnp + 8 + j) * BK + brow] = b1[j];
      }
    }
    if (wave == 0) __builtin_amdgcn_s_wait_tensorcnt(0);
    __syncthreads();

    v16bf af[4], bfv[2];
#pragma unroll
    for (int i = 0; i < 4; ++i) af[i] = frag_a(As, wm + i * 16, lane);
#pragma unroll
    for (int j = 0; j < 2; ++j) bfv[j] = frag_b(Bs, wn + j * 16, lane);
#pragma unroll
    for (int i = 0; i < 4; ++i)
#pragma unroll
      for (int j = 0; j < 2; ++j)
        acc[i][j] = __builtin_amdgcn_wmma_f32_16x16x32_bf16(
            false, af[i], false, bfv[j], (short)0, acc[i][j], false, false);
    __syncthreads();
  }

  const int half = lane >> 4, r = lane & 15;
#pragma unroll
  for (int i = 0; i < 4; ++i)
#pragma unroll
    for (int j = 0; j < 2; ++j)
#pragma unroll
      for (int e = 0; e < 8; ++e) {
        const int row = m0 + wm + i * 16 + 8 * half + e;
        const int col = n0 + wn + j * 16 + r;
        float v = acc[i][j][e];
        if (bias) v += bias[row];
        C[(size_t)row * ldc + col] = f2bf(v);
      }
}

// ------- C = scale * A(MxK, k-contig) @ B(NxK, k-contig)^T [+bias_m][+C] f32
__global__ __launch_bounds__(256)
void gemm_nt_f32(const bf16* __restrict__ A, long lda,
                 const bf16* __restrict__ B, long ldb,
                 float scale, const float* __restrict__ bias,
                 float* __restrict__ C, long ldc, int K, int accum)
{
  __shared__ bf16 As[BM * BK];
  __shared__ bf16 Bs[BN * BK];
  const int t = threadIdx.x, lane = t & 31, wave = t >> 5;
  const int m0 = blockIdx.y * BM, n0 = blockIdx.x * BN;
  const int wm = (wave & 1) * 64, wn = (wave >> 1) * 32;
  v8f acc[4][2] = {};

  const int row = t >> 1, kp = (t & 1) * 16;
  const unsigned ldsAs = lds_addr(As);
  const unsigned ldsB = lds_addr(Bs + row * BK + kp);

  for (int k0 = 0; k0 < K; k0 += BK) {
    // A tile via TDM (one descriptor / 8KB), B tile via per-lane async LDS
    if (wave == 0)
      tdm_load_tile(ldsAs, A + (size_t)m0 * lda + k0, lda, BM);
    async_copy_32B(ldsB, B + (size_t)(n0 + row) * ldb + k0 + kp);
    if (wave == 0) __builtin_amdgcn_s_wait_tensorcnt(0);
    wait_async();
    __syncthreads();

    v16bf af[4], bfv[2];
#pragma unroll
    for (int i = 0; i < 4; ++i) af[i] = frag_a(As, wm + i * 16, lane);
#pragma unroll
    for (int j = 0; j < 2; ++j) bfv[j] = frag_b(Bs, wn + j * 16, lane);
#pragma unroll
    for (int i = 0; i < 4; ++i)
#pragma unroll
      for (int j = 0; j < 2; ++j)
        acc[i][j] = __builtin_amdgcn_wmma_f32_16x16x32_bf16(
            false, af[i], false, bfv[j], (short)0, acc[i][j], false, false);
    __syncthreads();
  }

  const int half = lane >> 4, r = lane & 15;
#pragma unroll
  for (int i = 0; i < 4; ++i)
#pragma unroll
    for (int j = 0; j < 2; ++j)
#pragma unroll
      for (int e = 0; e < 8; ++e) {
        const int rowg = m0 + wm + i * 16 + 8 * half + e;
        const int colg = n0 + wn + j * 16 + r;
        const size_t idx = (size_t)rowg * ldc + colg;
        float v = acc[i][j][e] * scale;
        if (bias) v += bias[rowg];
        if (accum) v += C[idx];
        C[idx] = v;
      }
}

// ---------------- fp32 -> bf16 bulk convert (x4 vectorized)
__global__ __launch_bounds__(256)
void cvt_f32_bf16(const float* __restrict__ src, bf16* __restrict__ dst, long n4)
{
  long i = (long)blockIdx.x * blockDim.x + threadIdx.x;
  if (i < n4) {
    float4 f = ((const float4*)src)[i];
    v4bf o = { f2bf(f.x), f2bf(f.y), f2bf(f.z), f2bf(f.w) };
    ((v4bf*)dst)[i] = o;
  }
}

// ---------------- softmax over the HEAD axis (torch F.softmax dim=0 quirk)
__global__ __launch_bounds__(256)
void softmax_heads(const float* __restrict__ scores, bf16* __restrict__ attn)
{
  const size_t idx = (size_t)blockIdx.x * blockDim.x + threadIdx.x;
  const size_t ps = (size_t)L_ * L_;
  float v[H_];
  float mx = -INFINITY;
#pragma unroll
  for (int h = 0; h < H_; ++h) {
    v[h] = scores[(size_t)h * ps + idx];
    mx = fmaxf(mx, v[h]);
  }
  float s = 0.f;
#pragma unroll
  for (int h = 0; h < H_; ++h) { v[h] = __expf(v[h] - mx); s += v[h]; }
  const float inv = 1.f / s;
#pragma unroll
  for (int h = 0; h < H_; ++h)
    attn[(size_t)h * ps + idx] = f2bf(v[h] * inv);
}

// ---------------------------------------------------------------------------
extern "C" void kernel_launch(void* const* d_in, const int* in_sizes, int n_in,
                              void* d_out, int out_size, void* d_ws, size_t ws_size,
                              hipStream_t stream)
{
  (void)in_sizes; (void)n_in; (void)out_size; (void)ws_size;
  const float* tokens = (const float*)d_in[0];
  const float* Wq = (const float*)d_in[1];
  const float* bq = (const float*)d_in[2];
  const float* Wk = (const float*)d_in[3];
  const float* bk = (const float*)d_in[4];
  const float* Wv = (const float*)d_in[5];
  const float* bv = (const float*)d_in[6];
  const float* Wo = (const float*)d_in[7];
  const float* bo = (const float*)d_in[8];
  float* out = (float*)d_out;

  char* ws = (char*)d_ws;
  size_t off = 0;
  auto alloc = [&](size_t bytes) {
    char* p = ws + off;
    off += (bytes + 255) & ~(size_t)255;
    return p;
  };
  bf16*  tokb   = (bf16*)alloc((size_t)D_ * L_ * 2);      //   4 MB
  bf16*  wqb    = (bf16*)alloc((size_t)HD_ * D_ * 2);     //  16 MB
  bf16*  wkb    = (bf16*)alloc((size_t)HD_ * D_ * 2);
  bf16*  wvb    = (bf16*)alloc((size_t)HD_ * D_ * 2);
  bf16*  wob    = (bf16*)alloc((size_t)D_ * HD_ * 2);
  bf16*  qb     = (bf16*)alloc((size_t)HD_ * L_ * 2);     //  32 MB each
  bf16*  kb     = (bf16*)alloc((size_t)HD_ * L_ * 2);
  bf16*  vb     = (bf16*)alloc((size_t)HD_ * L_ * 2);
  float* scores = (float*)alloc((size_t)H_ * L_ * L_ * 4);// 128 MB
  bf16*  attnb  = (bf16*)alloc((size_t)H_ * L_ * L_ * 2); //  64 MB
  bf16*  ctxb   = (bf16*)alloc((size_t)H_ * L_ * D_ * 2); //  32 MB

  const dim3 blk(256);

  // 1) bf16 conversions
  cvt_f32_bf16<<<((size_t)D_ * L_ / 4) / 256, blk, 0, stream>>>(tokens, tokb, (long)D_ * L_ / 4);
  cvt_f32_bf16<<<((size_t)HD_ * D_ / 4) / 256, blk, 0, stream>>>(Wq, wqb, (long)HD_ * D_ / 4);
  cvt_f32_bf16<<<((size_t)HD_ * D_ / 4) / 256, blk, 0, stream>>>(Wk, wkb, (long)HD_ * D_ / 4);
  cvt_f32_bf16<<<((size_t)HD_ * D_ / 4) / 256, blk, 0, stream>>>(Wv, wvb, (long)HD_ * D_ / 4);
  cvt_f32_bf16<<<((size_t)D_ * HD_ / 4) / 256, blk, 0, stream>>>(Wo, wob, (long)D_ * HD_ / 4);

  // 2) QKV projections: q_flat = Wq @ tokens + bq  (8192x2048), bf16 out
  {
    dim3 grid(L_ / BN, HD_ / BM);
    gemm_nn_bf16<<<grid, blk, 0, stream>>>(wqb, D_, tokb, L_, bq, qb, L_, D_);
    gemm_nn_bf16<<<grid, blk, 0, stream>>>(wkb, D_, tokb, L_, bk, kb, L_, D_);
    gemm_nn_bf16<<<grid, blk, 0, stream>>>(wvb, D_, tokb, L_, bv, vb, L_, D_);
  }

  // 3) scores_h = (Q_h @ K_h^T) / 32, with split-heads view:
  //    Q_h[l,d] = q_flat[4l + h/2, (h&1)*1024 + d]  -> lda = 4*L_
  for (int h = 0; h < H_; ++h) {
    const bf16* qh = qb + (size_t)(h >> 1) * L_ + (size_t)(h & 1) * D_;
    const bf16* kh = kb + (size_t)(h >> 1) * L_ + (size_t)(h & 1) * D_;
    dim3 grid(L_ / BN, L_ / BM);
    gemm_nt_f32<<<grid, blk, 0, stream>>>(qh, 4L * L_, kh, 4L * L_, 0.03125f,
                                          nullptr, scores + (size_t)h * L_ * L_,
                                          L_, D_, 0);
  }

  // 4) softmax across heads (axis 0)
  softmax_heads<<<((size_t)L_ * L_) / 256, blk, 0, stream>>>(scores, attnb);

  // 5) ctx^T_h = attn_h (LxL) @ V_h (LxD)  -> bf16 (L x D) per head
  for (int h = 0; h < H_; ++h) {
    const bf16* vh = vb + (size_t)(h >> 1) * L_ + (size_t)(h & 1) * D_;
    dim3 grid(D_ / BN, L_ / BM);
    gemm_nn_bf16<<<grid, blk, 0, stream>>>(attnb + (size_t)h * L_ * L_, L_,
                                           vh, 4L * L_, nullptr,
                                           ctxb + (size_t)h * L_ * D_, D_, L_);
  }

  // 6) out = Wo @ ctx_flat + bo : accumulate 8 per-head NT GEMMs (f32 out)
  for (int h = 0; h < H_; ++h) {
    dim3 grid(L_ / BN, D_ / BM);
    gemm_nt_f32<<<grid, blk, 0, stream>>>(wob + (size_t)h * D_, HD_,
                                          ctxb + (size_t)h * L_ * D_, D_,
                                          1.0f, (h == 0) ? bo : nullptr,
                                          out, L_, D_, (h == 0) ? 0 : 1);
  }
}
